// SideChainDocker_19404662243784
// MI455X (gfx1250) — compile-verified, matrix operand order checked by reading, not run on hardware
//
#include <hip/hip_runtime.h>
#include <hip/hip_bf16.h>

// ---------------------------------------------------------------------------
// SideChainDocker for MI455X (gfx1250, wave32, WMMA)
//
// Pipeline (all on `stream`):
//   1. cvt_h       : h   f32 -> bf16                      (67 MB ws)
//   2. cvt_w1t     : W1  f32 -> bf16, transposed [n][k]   ( 8 MB ws)
//   3. gemm1_fused : X1 = relu(h@W1+b1) via v_wmma_f32_16x16x32_bf16,
//                    LDS double-buffered with GLOBAL_LOAD_ASYNC_TO_LDS_B128
//                    (ASYNCcnt); fused partial X1@W2 per N-tile
//                    -> angp[16][16384][14] f32           (14.7 MB ws)
//   4. geom_kernel : sum slices + b2, normalize angles, rigid-frame math,
//                    masked SE loss partials per block    (deterministic)
//   5. finalize    : reduce partials -> loss scalar
//
// Workspace requirement: ~90.2 MB.
// ---------------------------------------------------------------------------

typedef __bf16 bf16;
typedef __attribute__((ext_vector_type(16))) __bf16 v16bf;
typedef __attribute__((ext_vector_type(8)))  float  v8f;
typedef int v4i_gcc __attribute__((vector_size(16)));   // matches builtin proto

constexpr int MROWS = 8 * 2048;   // 16384 flattened (B*N) residues
constexpr int HD    = 2048;       // hidden dim
constexpr int NATOM = 14;
constexpr int BM = 128, BN = 128, BK = 32;
constexpr int MT = MROWS / BM;    // 128
constexpr int NT = HD / BN;       // 16
constexpr int KT = HD / BK;       // 64
constexpr int LDP = 40;           // LDS row pitch in bf16 elems (80 B, conflict-free)

#if __has_builtin(__builtin_amdgcn_global_load_async_to_lds_b128)
#define HAVE_ASYNC_LDS 1
#else
#define HAVE_ASYNC_LDS 0
#endif

__device__ __forceinline__ unsigned short f2bf(float f) {
  unsigned int u = __float_as_uint(f);
  u += 0x7FFFu + ((u >> 16) & 1u);      // round-to-nearest-even
  return (unsigned short)(u >> 16);
}
__device__ __forceinline__ float bf2f(unsigned short h) {
  return __uint_as_float(((unsigned int)h) << 16);
}

#if HAVE_ASYNC_LDS
// per-lane 16 B memory -> LDS async copy (tracked by ASYNCcnt)
__device__ __forceinline__ void async_cp16(const unsigned short* g, unsigned short* l) {
  __builtin_amdgcn_global_load_async_to_lds_b128(
      (__attribute__((address_space(1))) v4i_gcc*)g,
      (__attribute__((address_space(3))) v4i_gcc*)l, 0, 0);
}
#endif

__device__ __forceinline__ void wait_async0() {
#if __has_builtin(__builtin_amdgcn_s_wait_asynccnt)
  __builtin_amdgcn_s_wait_asynccnt(0);
#else
  asm volatile("s_wait_asynccnt 0x0" ::: "memory");
#endif
}

// ------------------------------ conversions --------------------------------

__global__ void cvt_h(const float* __restrict__ in, unsigned short* __restrict__ out,
                      long long n4) {
  long long i = (long long)blockIdx.x * blockDim.x + threadIdx.x;
  if (i >= n4) return;
  float4 v = ((const float4*)in)[i];
  ushort4 o;
  o.x = f2bf(v.x); o.y = f2bf(v.y); o.z = f2bf(v.z); o.w = f2bf(v.w);
  ((ushort4*)out)[i] = o;
}

// W1 [k][n] f32  ->  w1t [n][k] bf16  (LDS-tiled transpose)
__global__ void cvt_w1t(const float* __restrict__ W1, unsigned short* __restrict__ w1t) {
  __shared__ float t[32][33];
  const int n0 = blockIdx.x * 32, k0 = blockIdx.y * 32;
  const int tx = threadIdx.x, ty = threadIdx.y;  // 32 x 8
#pragma unroll
  for (int j = 0; j < 4; ++j)
    t[ty + j * 8][tx] = W1[(size_t)(k0 + ty + j * 8) * HD + n0 + tx];
  __syncthreads();
#pragma unroll
  for (int j = 0; j < 4; ++j)
    w1t[(size_t)(n0 + ty + j * 8) * HD + k0 + tx] = f2bf(t[tx][ty + j * 8]);
}

// ------------------------------ fused GEMM ---------------------------------

union GemmSmem {
  struct { unsigned short A[2][BM * LDP]; unsigned short B[2][BN * LDP]; } s; // 40 KB
  unsigned short X[BM * 132];                                                 // 33 KB
};
union Frag { uint4 q[2]; v16bf v; };

__global__ __launch_bounds__(256)
void gemm1_fused(const unsigned short* __restrict__ hbf,
                 const unsigned short* __restrict__ w1t,
                 const float* __restrict__ b1,
                 const float* __restrict__ W2,
                 float* __restrict__ angp) {
  __shared__ __align__(16) GemmSmem sm;
  const int tid  = threadIdx.x;
  const int lane = tid & 31;
  const int wid  = tid >> 5;
  const int wm   = wid >> 2;             // 0..1 : 64-row slab
  const int wn   = wid & 3;              // 0..3 : 32-col slab
  const int m0   = blockIdx.x * BM;
  const int n0   = blockIdx.y * BN;

  // staging: each thread copies two 16 B chunks of A and of B per K-step
  const int r0 = tid >> 2,          p0 = tid & 3;
  const int r1 = (tid + 256) >> 2,  p1 = (tid + 256) & 3;
  const unsigned short* gA0 = hbf + (size_t)(m0 + r0) * HD + p0 * 8;
  const unsigned short* gA1 = hbf + (size_t)(m0 + r1) * HD + p1 * 8;
  const unsigned short* gB0 = w1t + (size_t)(n0 + r0) * HD + p0 * 8;
  const unsigned short* gB1 = w1t + (size_t)(n0 + r1) * HD + p1 * 8;
  unsigned short* lA0 = &sm.s.A[0][r0 * LDP + p0 * 8];
  unsigned short* lA1 = &sm.s.A[0][r1 * LDP + p1 * 8];
  unsigned short* lB0 = &sm.s.B[0][r0 * LDP + p0 * 8];
  unsigned short* lB1 = &sm.s.B[0][r1 * LDP + p1 * 8];
  const int bufStride = BM * LDP;        // elems between buffer 0 and 1

  v8f acc[4][2];
#pragma unroll
  for (int s = 0; s < 4; ++s)
#pragma unroll
    for (int t = 0; t < 2; ++t) acc[s][t] = v8f{};

  // ---- stage first K tile into buffer 0 ----
#if HAVE_ASYNC_LDS
  async_cp16(gA0, lA0);
  async_cp16(gA1, lA1);
  async_cp16(gB0, lB0);
  async_cp16(gB1, lB1);
  wait_async0();
#else
  {
    uint4 a0 = *(const uint4*)gA0, a1 = *(const uint4*)gA1;
    uint4 b0 = *(const uint4*)gB0, b1v = *(const uint4*)gB1;
    *(uint4*)lA0 = a0;
    *(uint4*)lA1 = a1;
    *(uint4*)lB0 = b0;
    *(uint4*)lB1 = b1v;
  }
#endif
  __syncthreads();

  const int lrow = lane & 15;
  const int koff = (lane >> 4) * 8;      // K half select per WMMA 16-bit A/B layout

  for (int kt = 0; kt < KT; ++kt) {
    const int  cur = kt & 1;
    const bool nx  = (kt + 1 < KT);

#if HAVE_ASYNC_LDS
    if (nx) {  // kick next tile straight into the other LDS buffer (ASYNCcnt)
      const int kk = (kt + 1) * BK;
      const int bo = (cur ^ 1) * bufStride;
      async_cp16(gA0 + kk, lA0 + bo);
      async_cp16(gA1 + kk, lA1 + bo);
      async_cp16(gB0 + kk, lB0 + bo);
      async_cp16(gB1 + kk, lB1 + bo);
    }
#else
    uint4 a0, a1, b0, b1v;
    if (nx) {
      const int kk = (kt + 1) * BK;
      a0  = *(const uint4*)(gA0 + kk);
      a1  = *(const uint4*)(gA1 + kk);
      b0  = *(const uint4*)(gB0 + kk);
      b1v = *(const uint4*)(gB1 + kk);
    }
#endif

    Frag bf[2];
#pragma unroll
    for (int t = 0; t < 2; ++t) {
      const unsigned short* p = &sm.s.B[cur][(wn * 32 + t * 16 + lrow) * LDP + koff];
      bf[t].q[0] = *(const uint4*)p;
      bf[t].q[1] = *(const uint4*)(p + 16);
    }
#pragma unroll
    for (int s = 0; s < 4; ++s) {
      Frag af;
      const unsigned short* p = &sm.s.A[cur][(wm * 64 + s * 16 + lrow) * LDP + koff];
      af.q[0] = *(const uint4*)p;
      af.q[1] = *(const uint4*)(p + 16);
#pragma unroll
      for (int t = 0; t < 2; ++t)
        acc[s][t] = __builtin_amdgcn_wmma_f32_16x16x32_bf16(
            false, af.v, false, bf[t].v, (short)0, acc[s][t], false, false);
    }

#if HAVE_ASYNC_LDS
    if (nx) wait_async0();
#else
    if (nx) {
      const int bo = (cur ^ 1) * bufStride;
      *(uint4*)(lA0 + bo) = a0;
      *(uint4*)(lA1 + bo) = a1;
      *(uint4*)(lB0 + bo) = b0;
      *(uint4*)(lB1 + bo) = b1v;
    }
#endif
    __syncthreads();
  }

  // epilogue: bias + relu -> bf16 tile in LDS (C layout: vgpr r = row, lane hi half = +8)
#pragma unroll
  for (int t = 0; t < 2; ++t) {
    const int   nc   = wn * 32 + t * 16 + lrow;
    const float bias = b1[n0 + nc];
#pragma unroll
    for (int s = 0; s < 4; ++s)
#pragma unroll
      for (int r = 0; r < 8; ++r) {
        const int mr = wm * 64 + s * 16 + (lane >> 4) * 8 + r;
        float x = acc[s][t][r] + bias;
        sm.X[mr * 132 + nc] = f2bf(fmaxf(x, 0.f));
      }
  }
  __syncthreads();

  // fused partial GEMM2: tile(128x128) @ W2[n0..n0+127][0..13] -> angp slice
  const int mr = tid & 127;
  const int ch = tid >> 7;               // 0/1 -> cols ch*7 .. ch*7+6
  float pacc[7];
#pragma unroll
  for (int c = 0; c < 7; ++c) pacc[c] = 0.f;
  for (int k = 0; k < BN; ++k) {
    const float xv = bf2f(sm.X[mr * 132 + k]);
    const float* wp = W2 + (size_t)(n0 + k) * NATOM + ch * 7;
#pragma unroll
    for (int c = 0; c < 7; ++c) pacc[c] += xv * wp[c];
  }
  float* op = angp + (size_t)blockIdx.y * (MROWS * NATOM) + (size_t)(m0 + mr) * NATOM + ch * 7;
#pragma unroll
  for (int c = 0; c < 7; ++c) op[c] = pacc[c];
}

// ------------------------------ geometry -----------------------------------

__device__ __forceinline__ void mm3(const float* A, const float* B, float* C) {
#pragma unroll
  for (int r = 0; r < 3; ++r)
#pragma unroll
    for (int c = 0; c < 3; ++c)
      C[r * 3 + c] = A[r * 3 + 0] * B[0 * 3 + c] + A[r * 3 + 1] * B[1 * 3 + c] +
                     A[r * 3 + 2] * B[2 * 3 + c];
}
__device__ __forceinline__ void mv3(const float* A, const float* v, float* o) {
#pragma unroll
  for (int r = 0; r < 3; ++r)
    o[r] = A[r * 3 + 0] * v[0] + A[r * 3 + 1] * v[1] + A[r * 3 + 2] * v[2];
}

__global__ __launch_bounds__(128)
void geom_kernel(const float* __restrict__ angp, const float* __restrict__ b2,
                 const float* __restrict__ true_X, const int* __restrict__ bind_A,
                 const int* __restrict__ bind_aa, const float* __restrict__ dframes,
                 const int* __restrict__ gidx, const float* __restrict__ amask,
                 const float* __restrict__ litpos, float2* __restrict__ lossp) {
  __shared__ float  fr[128 * 97];        // 8 frames x (R 9 + t 3) per thread, pad 97
  __shared__ float2 red[128];
  const int tid = threadIdx.x;
  const int i   = blockIdx.x * 128 + tid;
  float* myf = &fr[tid * 97];

  // sum GEMM2 partial slices + bias
  float x14[14];
#pragma unroll
  for (int c = 0; c < 14; ++c) x14[c] = b2[c];
  for (int s = 0; s < NT; ++s)
#pragma unroll
    for (int c = 0; c < 14; ++c)
      x14[c] += angp[(size_t)s * (MROWS * NATOM) + (size_t)i * NATOM + c];

  // normalized (sin, cos) pairs; frame 0 = identity twist
  float s8[8], c8[8];
  s8[0] = 0.f; c8[0] = 1.f;
#pragma unroll
  for (int j = 0; j < 7; ++j) {
    float s = x14[2 * j], c = x14[2 * j + 1];
    float n = sqrtf(s * s + c * c);
    n = fmaxf(n, 1e-12f);
    s8[j + 1] = s / n; c8[j + 1] = c / n;
  }

  // backbone rigid from atoms 0,1,2
  const float* Xr = true_X + (size_t)i * (NATOM * 3);
  float e0[3], e1[3], e2[3], tb[3];
#pragma unroll
  for (int r = 0; r < 3; ++r) {
    e0[r] = Xr[3 + r] - Xr[r];
    e1[r] = Xr[6 + r] - Xr[3 + r];
    tb[r] = Xr[3 + r];
  }
  float inv0 = rsqrtf(e0[0]*e0[0] + e0[1]*e0[1] + e0[2]*e0[2] + 1e-8f);
#pragma unroll
  for (int r = 0; r < 3; ++r) e0[r] *= inv0;
  float dp = e0[0]*e1[0] + e0[1]*e1[1] + e0[2]*e1[2];
#pragma unroll
  for (int r = 0; r < 3; ++r) e1[r] -= e0[r] * dp;
  float inv1 = rsqrtf(e1[0]*e1[0] + e1[1]*e1[1] + e1[2]*e1[2] + 1e-8f);
#pragma unroll
  for (int r = 0; r < 3; ++r) e1[r] *= inv1;
  e2[0] = e0[1]*e1[2] - e0[2]*e1[1];
  e2[1] = e0[2]*e1[0] - e0[0]*e1[2];
  e2[2] = e0[0]*e1[1] - e0[1]*e1[0];
  const float Rb[9] = { e0[0], e1[0], e2[0],
                        e0[1], e1[1], e2[1],
                        e0[2], e1[2], e2[2] };

  const int aa  = bind_aa[i];
  const int aat = max(aa - 1, 0);
  const float* dfp = dframes + (size_t)aat * 128;   // [8][4][4]

  float Rcur[9], tcur[3];
#pragma unroll
  for (int f = 0; f < 8; ++f) {
    // Rf = Rd @ Rx(s,c), tf = td
    float Rf_[9], tf_[3];
    {
      const float* d = dfp + f * 16;
      const float s = s8[f], c = c8[f];
#pragma unroll
      for (int r = 0; r < 3; ++r) {
        float d1 = d[r * 4 + 1], d2 = d[r * 4 + 2];
        Rf_[r * 3 + 0] = d[r * 4 + 0];
        Rf_[r * 3 + 1] =  c * d1 + s * d2;
        Rf_[r * 3 + 2] = -s * d1 + c * d2;
        tf_[r] = d[r * 4 + 3];
      }
    }
    float Ra[9], ta[3];
    if (f < 5) {
#pragma unroll
      for (int j = 0; j < 9; ++j) Ra[j] = Rf_[j];
#pragma unroll
      for (int r = 0; r < 3; ++r) ta[r] = tf_[r];
      if (f == 4) {
#pragma unroll
        for (int j = 0; j < 9; ++j) Rcur[j] = Rf_[j];
#pragma unroll
        for (int r = 0; r < 3; ++r) tcur[r] = tf_[r];
      }
    } else {
      float tn[3], Rn[9];
      mv3(Rcur, tf_, tn);
#pragma unroll
      for (int r = 0; r < 3; ++r) tn[r] += tcur[r];
      mm3(Rcur, Rf_, Rn);
#pragma unroll
      for (int j = 0; j < 9; ++j) { Rcur[j] = Rn[j]; Ra[j] = Rn[j]; }
#pragma unroll
      for (int r = 0; r < 3; ++r) { tcur[r] = tn[r]; ta[r] = tn[r]; }
    }
    // global frame: Rg = Rbb @ Ra, tg = Rbb @ ta + tbb -> LDS
    float Rg[9], tg[3];
    mm3(Rb, Ra, Rg);
    mv3(Rb, ta, tg);
#pragma unroll
    for (int j = 0; j < 9; ++j) myf[f * 12 + j] = Rg[j];
#pragma unroll
    for (int r = 0; r < 3; ++r) myf[f * 12 + 9 + r] = tg[r] + tb[r];
  }

  // atoms
  const int* bA = bind_A + (size_t)i * NATOM;
  float sum_se = 0.f, sum_m = 0.f;
#pragma unroll
  for (int a = 0; a < NATOM; ++a) {
    const int g = gidx[aat * NATOM + a];
    const float* R = &myf[g * 12];
    const float px = litpos[(aat * NATOM + a) * 3 + 0];
    const float py = litpos[(aat * NATOM + a) * 3 + 1];
    const float pz = litpos[(aat * NATOM + a) * 3 + 2];
    float pos0 = R[0]*px + R[1]*py + R[2]*pz + R[9];
    float pos1 = R[3]*px + R[4]*py + R[5]*pz + R[10];
    float pos2 = R[6]*px + R[7]*py + R[8]*pz + R[11];
    const float bm = (bA[a] > 0) ? 1.f : 0.f;
    const float mk = amask[aat * NATOM + a] * bm;
    pos0 *= mk; pos1 *= mk; pos2 *= mk;
    if (a >= 4) {
      float d0 = pos0 - Xr[a * 3 + 0];
      float d1 = pos1 - Xr[a * 3 + 1];
      float d2 = pos2 - Xr[a * 3 + 2];
      sum_se += (d0*d0 + d1*d1 + d2*d2) * bm;
      sum_m  += bm;
    }
  }

  red[tid] = make_float2(sum_se, sum_m);
  __syncthreads();
  for (int s = 64; s > 0; s >>= 1) {
    if (tid < s) { red[tid].x += red[tid + s].x; red[tid].y += red[tid + s].y; }
    __syncthreads();
  }
  if (tid == 0) lossp[blockIdx.x] = red[0];
}

__global__ void finalize(const float2* __restrict__ lossp, float* __restrict__ out) {
  __shared__ float2 red[128];
  const int t = threadIdx.x;
  red[t] = lossp[t];
  __syncthreads();
  for (int s = 64; s > 0; s >>= 1) {
    if (t < s) { red[t].x += red[t + s].x; red[t].y += red[t + s].y; }
    __syncthreads();
  }
  if (t == 0) out[0] = red[0].x / fmaxf(red[0].y, 1e-4f);
}

// ------------------------------ launcher -----------------------------------

extern "C" void kernel_launch(void* const* d_in, const int* in_sizes, int n_in,
                              void* d_out, int out_size, void* d_ws, size_t ws_size,
                              hipStream_t stream) {
  (void)in_sizes; (void)n_in; (void)out_size;
  const float* h       = (const float*)d_in[0];
  const float* true_X  = (const float*)d_in[1];
  const int*   bind_A  = (const int*)  d_in[2];
  const int*   bind_aa = (const int*)  d_in[3];
  const float* W1      = (const float*)d_in[4];
  const float* b1      = (const float*)d_in[5];
  const float* W2      = (const float*)d_in[6];
  const float* b2      = (const float*)d_in[7];
  const float* dframes = (const float*)d_in[8];
  const int*   gidx    = (const int*)  d_in[9];
  const float* amask   = (const float*)d_in[10];
  const float* litpos  = (const float*)d_in[11];
  float* out = (float*)d_out;

  // workspace carve (needs ~90.2 MB)
  char* w = (char*)d_ws;
  unsigned short* hbf  = (unsigned short*)w; w += (size_t)MROWS * HD * 2;         // 64 MB
  unsigned short* w1t  = (unsigned short*)w; w += (size_t)HD * HD * 2;            //  8 MB
  float*          angp = (float*)w;          w += (size_t)NT * MROWS * NATOM * 4; // 14 MB
  float2*         lossp = (float2*)w;
  (void)ws_size;

  const long long n4 = (long long)MROWS * HD / 4;
  cvt_h<<<(unsigned)((n4 + 255) / 256), 256, 0, stream>>>(h, hbf, n4);
  cvt_w1t<<<dim3(HD / 32, HD / 32), dim3(32, 8), 0, stream>>>(W1, w1t);
  gemm1_fused<<<dim3(MT, NT), 256, 0, stream>>>(hbf, w1t, b1, W2, angp);
  geom_kernel<<<MROWS / 128, 128, 0, stream>>>(angp, b2, true_X, bind_A, bind_aa,
                                               dframes, gidx, amask, litpos, lossp);
  finalize<<<1, 128, 0, stream>>>(lossp, out);
}
